// TargetQuerySourceBottleneck_44684839748409
// MI455X (gfx1250) — compile-verified, matrix operand order checked by reading, not run on hardware
//
#include <hip/hip_runtime.h>
#include <hip/hip_bf16.h>
#include <math.h>

typedef unsigned short u16;
typedef unsigned int   u32;
typedef __bf16 bf16_t;
typedef bf16_t v16bf __attribute__((ext_vector_type(16)));
typedef float  v8f   __attribute__((ext_vector_type(8)));
typedef u32    u32x4 __attribute__((ext_vector_type(4)));

union Frag { v16bf v; u32x4 q[2]; };

__device__ __forceinline__ u16 f2bf(float f) {
  u32 u = __builtin_bit_cast(u32, f);
  u32 r = u + 0x7FFFu + ((u >> 16) & 1u);   // round-to-nearest-even
  return (u16)(r >> 16);
}
__device__ __forceinline__ float bf2f(u16 h) {
  return __builtin_bit_cast(float, (u32)h << 16);
}
__device__ __forceinline__ v8f zero8() {
  v8f z; for (int i = 0; i < 8; ++i) z[i] = 0.f; return z;
}
// Inline tanh via hardware v_exp_f32 (avoids OCML call -> no VGPR spills).
// Clamp keeps e^{2x} finite; |x|>=15 gives +/-1 to well past bf16 precision.
__device__ __forceinline__ float tanh_fast(float x) {
  float xc = fminf(fmaxf(x, -15.f), 15.f);
  float t  = __expf(2.f * xc);
  return (t - 1.f) / (t + 1.f);
}

// ---------------------------------------------------------------------------
// Kernel 1: hidden = tanh(X @ W + b), fp32 in, bf16 out.  X:[M,4096] W:[4096,1024]
// 128x128 tile per WG, 8 waves (2Mx4N), BK=64, bf16 staged in LDS.
// ---------------------------------------------------------------------------
#define KD 4096
#define ND 1024
#define LDA 72   // padded LDS row stride in halves (144B, 16B aligned)

__global__ __launch_bounds__(256)
void gemm_tanh_bf16(const float* __restrict__ X, const float* __restrict__ W,
                    const float* __restrict__ bias, u16* __restrict__ out, int M) {
  extern __shared__ char smem[];
  u16* As = (u16*)smem;                     // [128][LDA]  row=m, col=k
  u16* Bs = (u16*)(smem + 128 * LDA * 2);   // [128][LDA]  row=h, col=k (transposed)

  const int tid  = threadIdx.x;
  const int lane = tid & 31;
  const int wave = tid >> 5;
  const int m0 = blockIdx.x * 128;
  const int n0 = blockIdx.y * 128;
  const int wm = (wave & 1) * 64;
  const int wn = (wave >> 1) * 32;

  v8f acc[4][2];
  for (int i = 0; i < 4; ++i) for (int j = 0; j < 2; ++j) acc[i][j] = zero8();

  const int lrow = lane & 15;
  const int klo  = (lane >> 4) << 3;   // 0 or 8 (wave32 half-lane K split)

  for (int kb = 0; kb < KD; kb += 64) {
    // A tile: X[m0..+128][kb..+64] -> bf16 LDS
    for (int it = 0; it < 8; ++it) {
      int idx = tid + it * 256;                  // 2048 float4 loads
      int r = idx >> 4;
      int c = (idx & 15) << 2;
      const float4 v = *(const float4*)(X + (size_t)(m0 + r) * KD + kb + c);
      u32 lo = (u32)f2bf(v.x) | ((u32)f2bf(v.y) << 16);
      u32 hi = (u32)f2bf(v.z) | ((u32)f2bf(v.w) << 16);
      u32* p = (u32*)(As + r * LDA + c);
      p[0] = lo; p[1] = hi;
    }
    // B tile transposed: W[kb+kk][n0+h] -> Bs[h][kk]
    for (int it = 0; it < 8; ++it) {
      int idx = tid + it * 256;
      int kk = idx >> 5;
      int h  = (idx & 31) << 2;
      const float4 v = *(const float4*)(W + (size_t)(kb + kk) * ND + n0 + h);
      Bs[(h + 0) * LDA + kk] = f2bf(v.x);
      Bs[(h + 1) * LDA + kk] = f2bf(v.y);
      Bs[(h + 2) * LDA + kk] = f2bf(v.z);
      Bs[(h + 3) * LDA + kk] = f2bf(v.w);
    }
    __syncthreads();

    for (int kk = 0; kk < 64; kk += 32) {
      Frag a[4], bf[2];
      for (int i = 0; i < 4; ++i) {
        const u16* p = As + (wm + i * 16 + lrow) * LDA + kk + klo;
        a[i].q[0] = *(const u32x4*)p;
        a[i].q[1] = *(const u32x4*)(p + 16);
      }
      for (int j = 0; j < 2; ++j) {
        const u16* p = Bs + (wn + j * 16 + lrow) * LDA + kk + klo;
        bf[j].q[0] = *(const u32x4*)p;
        bf[j].q[1] = *(const u32x4*)(p + 16);
      }
      for (int i = 0; i < 4; ++i)
        for (int j = 0; j < 2; ++j)
          acc[i][j] = __builtin_amdgcn_wmma_f32_16x16x32_bf16(
              false, a[i].v, false, bf[j].v, (short)0, acc[i][j], false, false);
    }
    __syncthreads();
  }

  // epilogue: bias + tanh (inline), store bf16.
  const int rhi = (lane >> 4) << 3;
  for (int i = 0; i < 4; ++i)
    for (int j = 0; j < 2; ++j) {
      int col = n0 + wn + j * 16 + lrow;
      float bv = bias[col];
      for (int e = 0; e < 8; ++e) {
        int row = m0 + wm + i * 16 + e + rhi;
        out[(size_t)row * ND + col] = f2bf(tanh_fast(acc[i][j][e] + bv));
      }
    }
}

// ---------------------------------------------------------------------------
// Kernel 2a: target summary = masked mean over TT
// ---------------------------------------------------------------------------
__global__ __launch_bounds__(256)
void target_summary_kernel(const u16* __restrict__ tgtH,
                           const unsigned char* __restrict__ tmask,
                           float* __restrict__ summary) {
  __shared__ float red[256];
  const int b = blockIdx.y;
  const int h = blockIdx.x * 256 + threadIdx.x;
  float mc = 0.f;
  for (int t = threadIdx.x; t < 512; t += 256) mc += tmask[b * 512 + t] ? 1.f : 0.f;
  red[threadIdx.x] = mc;
  __syncthreads();
  for (int s = 128; s > 0; s >>= 1) {
    if (threadIdx.x < s) red[threadIdx.x] += red[threadIdx.x + s];
    __syncthreads();
  }
  const float denom = fmaxf(red[0], 1.f);
  float sum = 0.f;
  for (int t = 0; t < 512; ++t) {
    float m = tmask[b * 512 + t] ? 1.f : 0.f;
    sum += m * bf2f(tgtH[((size_t)b * 512 + t) * 1024 + h]);
  }
  summary[b * 1024 + h] = sum / denom;
}

// ---------------------------------------------------------------------------
// Kernel 2b: q[b,i,h] = bf16(queries[i,h] + summary[b,h])
// ---------------------------------------------------------------------------
__global__ __launch_bounds__(256)
void build_q_kernel(const float* __restrict__ queries,
                    const float* __restrict__ summary, u16* __restrict__ qbuf) {
  int idx = blockIdx.x * 256 + threadIdx.x;   // 16*64*1024 total
  int h = idx & 1023;
  int i = (idx >> 10) & 63;
  int b = idx >> 16;
  qbuf[idx] = f2bf(queries[i * 1024 + h] + summary[b * 1024 + h]);
}

// ---------------------------------------------------------------------------
// Kernel 3: attention.  One WG per (batch b, 16-query tile).
// Pass1: scores = q @ srcH^T / 32, mask, softmax -> bf16 weights in LDS.
// Pass2: pooled partial = weights @ srcH via WMMA, atomicAdd column sums.
// ---------------------------------------------------------------------------
#define LDQ 1032                               // halves
#define LDW 2056                               // halves
#define LDH 40                                 // halves
#define SC_OFF  (16 * LDQ * 2)                 // 33024
#define WB_OFF  (SC_OFF + 16 * 2048 * 4)       // 164096
#define HT_OFF  (WB_OFF + 16 * LDW * 2)        // 229888
#define SMEM3   (HT_OFF + 1024 * LDH * 2)      // 311808 (< 320KB WGP LDS)

__global__ __launch_bounds__(256)
void attention_kernel(const u16* __restrict__ srcH,
                      const unsigned char* __restrict__ smask,
                      const u16* __restrict__ qbuf, float* __restrict__ pooled) {
  extern __shared__ char smem[];
  u16*   qs = (u16*)smem;               // [16][LDQ]
  float* sc = (float*)(smem + SC_OFF);  // [16][2048]
  u16*   wb = (u16*)(smem + WB_OFF);    // [16][LDW]
  u16*   ht = (u16*)(smem + HT_OFF);    // [1024][LDH] (hidden transposed chunk)

  const int tid  = threadIdx.x;
  const int lane = tid & 31;
  const int wave = tid >> 5;
  const int b  = blockIdx.y;
  const int q0 = blockIdx.x * 16;
  const int lrow = lane & 15;
  const int klo  = (lane >> 4) << 3;

  // stage q tile
  {
    const u32* src = (const u32*)(qbuf + ((size_t)b * 64 + q0) * 1024);
    u32* dst = (u32*)qs;
    for (int it = 0; it < 32; ++it) {
      int idx = tid + it * 256;      // 8192 u32
      int m = idx >> 9, c = idx & 511;
      dst[m * (LDQ / 2) + c] = src[m * 512 + c];
    }
  }
  __syncthreads();

  // ---- pass 1: scores ----
  const float scale = 0.03125f;  // 1/sqrt(1024)
  for (int tt = wave; tt < 128; tt += 8) {
    int tbase = tt * 16;
    v8f acc = zero8();
    const u16* hrow = srcH + ((size_t)b * 2048 + tbase + lrow) * 1024;
    for (int k = 0; k < 1024; k += 32) {
      Frag a, bb;
      const u16* pa = qs + lrow * LDQ + k + klo;
      a.q[0] = *(const u32x4*)pa;
      a.q[1] = *(const u32x4*)(pa + 16);
      const u16* pb = hrow + k + klo;              // contiguous per lane -> b128
      bb.q[0] = *(const u32x4*)pb;
      bb.q[1] = *(const u32x4*)(pb + 16);
      acc = __builtin_amdgcn_wmma_f32_16x16x32_bf16(
          false, a.v, false, bb.v, (short)0, acc, false, false);
    }
    int t = tbase + lrow;
    bool keep = smask[b * 2048 + t] != 0;
    int rhi = (lane >> 4) << 3;
    for (int e = 0; e < 8; ++e) {
      int m = e + rhi;
      sc[m * 2048 + t] = keep ? acc[e] * scale : -1e9f;
    }
  }
  __syncthreads();

  // ---- softmax over t (2 rows per wave) ----
  for (int r = wave * 2; r < wave * 2 + 2; ++r) {
    float mx = -1e30f;
    for (int t = lane; t < 2048; t += 32) mx = fmaxf(mx, sc[r * 2048 + t]);
    for (int s = 16; s > 0; s >>= 1) mx = fmaxf(mx, __shfl_xor(mx, s, 32));
    float sum = 0.f;
    for (int t = lane; t < 2048; t += 32) sum += __expf(sc[r * 2048 + t] - mx);
    for (int s = 16; s > 0; s >>= 1) sum += __shfl_xor(sum, s, 32);
    float inv = 1.f / sum;
    for (int t = lane; t < 2048; t += 32)
      wb[r * LDW + t] = f2bf(__expf(sc[r * 2048 + t] - mx) * inv);
  }
  __syncthreads();

  // ---- pass 2: pooled partial = weights @ hidden ----
  v8f acc2[8];
  for (int i = 0; i < 8; ++i) acc2[i] = zero8();
  for (int tc = 0; tc < 2048; tc += 32) {
    // stage hidden[tc..tc+32][0..1024] transposed into ht[h][t]
    const u32* src = (const u32*)(srcH + ((size_t)b * 2048 + tc) * 1024);
    for (int it = 0; it < 64; ++it) {
      int idx = tid + it * 256;    // 16384 u32
      int tl = idx >> 9, h2 = idx & 511;
      u32 v = src[tl * 512 + h2];
      ht[(h2 * 2 + 0) * LDH + tl] = (u16)(v & 0xFFFFu);
      ht[(h2 * 2 + 1) * LDH + tl] = (u16)(v >> 16);
    }
    __syncthreads();
    Frag a;
    const u16* pa = wb + lrow * LDW + tc + klo;
    a.q[0] = *(const u32x4*)pa;
    a.q[1] = *(const u32x4*)(pa + 16);
    for (int hs = 0; hs < 8; ++hs) {
      int hbase = wave * 128 + hs * 16;
      Frag bb;
      const u16* pb = ht + (hbase + lrow) * LDH + klo;
      bb.q[0] = *(const u32x4*)pb;
      bb.q[1] = *(const u32x4*)(pb + 16);
      acc2[hs] = __builtin_amdgcn_wmma_f32_16x16x32_bf16(
          false, a.v, false, bb.v, (short)0, acc2[hs], false, false);
    }
    __syncthreads();
  }
  // column sums over the 16 query rows -> atomic accumulate
  for (int hs = 0; hs < 8; ++hs) {
    float s = 0.f;
    for (int e = 0; e < 8; ++e) s += acc2[hs][e];
    s += __shfl_xor(s, 16, 32);
    if (lane < 16)
      atomicAdd(&pooled[b * 1024 + wave * 128 + hs * 16 + lane], s);
  }
}

// ---------------------------------------------------------------------------
// Kernel 4: heads.  out[272] = (pooled/Q) @ head_w_m + head_b_m, m in {2,3,5,7}
// ---------------------------------------------------------------------------
__global__ void heads_kernel(const float* __restrict__ pooled,
                             const float* __restrict__ w2, const float* __restrict__ b2,
                             const float* __restrict__ w3, const float* __restrict__ b3,
                             const float* __restrict__ w5, const float* __restrict__ b5,
                             const float* __restrict__ w7, const float* __restrict__ b7,
                             float* __restrict__ out) {
  int idx = blockIdx.x * 64 + threadIdx.x;
  if (idx >= 272) return;
  const float *w, *bias; int m, r;
  if (idx < 32)       { w = w2; bias = b2; m = 2; r = idx; }
  else if (idx < 80)  { w = w3; bias = b3; m = 3; r = idx - 32; }
  else if (idx < 160) { w = w5; bias = b5; m = 5; r = idx - 80; }
  else                { w = w7; bias = b7; m = 7; r = idx - 160; }
  int bidx = r / m, j = r % m;
  const float invQ = 1.f / 64.f;
  float s = 0.f;
  for (int h = 0; h < 1024; ++h)
    s += (pooled[bidx * 1024 + h] * invQ) * w[h * m + j];
  out[idx] = s + bias[j];
}

// ---------------------------------------------------------------------------
extern "C" void kernel_launch(void* const* d_in, const int* in_sizes, int n_in,
                              void* d_out, int out_size, void* d_ws, size_t ws_size,
                              hipStream_t stream) {
  const float* src_tok = (const float*)d_in[0];
  const unsigned char* src_mask = (const unsigned char*)d_in[1];
  const float* tgt_tok = (const float*)d_in[2];
  const unsigned char* tgt_mask = (const unsigned char*)d_in[3];
  const float* Ws = (const float*)d_in[4];
  const float* bs = (const float*)d_in[5];
  const float* Wt = (const float*)d_in[6];
  const float* bt = (const float*)d_in[7];
  const float* queries = (const float*)d_in[8];
  const float* hw2 = (const float*)d_in[9],  *hb2 = (const float*)d_in[10];
  const float* hw3 = (const float*)d_in[11], *hb3 = (const float*)d_in[12];
  const float* hw5 = (const float*)d_in[13], *hb5 = (const float*)d_in[14];
  const float* hw7 = (const float*)d_in[15], *hb7 = (const float*)d_in[16];

  // workspace carve (bytes)
  char* ws = (char*)d_ws;
  u16*   srcH    = (u16*)(ws + 0);             // 16*2048*1024*2 = 67,108,864
  u16*   tgtH    = (u16*)(ws + 67108864);      // 16*512*1024*2  = 16,777,216
  float* summary = (float*)(ws + 83886080);    // 16*1024*4      = 65,536
  u16*   qbuf    = (u16*)(ws + 83951616);      // 16*64*1024*2   = 2,097,152
  float* pooled  = (float*)(ws + 86048768);    // 16*1024*4      = 65,536

  const size_t smem1 = (size_t)2 * 128 * LDA * 2;   // 36,864 B

  // 1) bottleneck GEMMs (bf16 WMMA, tanh epilogue)
  gemm_tanh_bf16<<<dim3(32768 / 128, 1024 / 128), 256, smem1, stream>>>(
      src_tok, Ws, bs, srcH, 32768);
  gemm_tanh_bf16<<<dim3(8192 / 128, 1024 / 128), 256, smem1, stream>>>(
      tgt_tok, Wt, bt, tgtH, 8192);

  // 2) target summary + query build
  target_summary_kernel<<<dim3(4, 16), 256, 0, stream>>>(tgtH, tgt_mask, summary);
  build_q_kernel<<<4096, 256, 0, stream>>>(queries, summary, qbuf);

  // 3) attention (zero pooled accumulator first — deterministic per call)
  hipMemsetAsync(pooled, 0, 16 * 1024 * sizeof(float), stream);
  attention_kernel<<<dim3(4, 16), 256, SMEM3, stream>>>(srcH, src_mask, qbuf, pooled);

  // 4) modulus heads
  heads_kernel<<<dim3(5), 64, 0, stream>>>(pooled, hw2, hb2, hw3, hb3,
                                           hw5, hb5, hw7, hb7, (float*)d_out);
}
